// Geometry3DAnalyzer_11553462026408
// MI455X (gfx1250) — matrix operand order repaired and checked by reading, not compiled, not run on hardware
//
#include <hip/hip_runtime.h>
#include <hip/hip_bf16.h>

// ---------- types ----------
typedef __attribute__((ext_vector_type(16))) __bf16 v16bf;
typedef __attribute__((ext_vector_type(8)))  float  v8f;

#define B_ROWS   32768
#define D_IN     1404      // 468*3
#define K1PAD    1408      // 44 k-tiles
#define N1       256
#define N2       128
#define N3       64
#define KCOMB    96        // 64 lm + 6 pose + 26 zero pad
#define NF       512
#define NCLS     2
#define LDS_STRIDE 40      // shorts per row: 80B, keeps ds_*_b128 aligned + conflict-free gather

__device__ __forceinline__ unsigned short f2bf(float f) {
    unsigned int u = __builtin_bit_cast(unsigned int, f);
    u += 0x7FFFu + ((u >> 16) & 1u);          // round-to-nearest-even
    return (unsigned short)(u >> 16);
}

__device__ __forceinline__ unsigned int pack2bf(float a, float b) {
    return (unsigned int)f2bf(a) | ((unsigned int)f2bf(b) << 16);
}

// ---------- weight packing into WMMA B-fragment layout ----------
// dst[((kt*NT + nt)*32 + lane)*8 + v] = bf16 pair (K = kt*32 + 2v + 16*half, +1), col = nt*16 + lane%16
__global__ void pack_w_kernel(const float* __restrict__ W, int Ksrc, int Ncols,
                              int KT, int NT, unsigned int* __restrict__ dst) {
    int tid   = blockIdx.x * blockDim.x + threadIdx.x;
    int total = KT * NT * 256;
    if (tid >= total) return;
    int v     = tid & 7;
    int lane  = (tid >> 3) & 31;
    int fragi = tid >> 8;
    int nt    = fragi % NT;
    int kt    = fragi / NT;
    int half  = lane >> 4;
    int col   = nt * 16 + (lane & 15);
    int k0    = kt * 32 + 2 * v + 16 * half;
    float w0  = (k0     < Ksrc) ? W[(size_t)k0 * Ncols + col]       : 0.f;
    float w1  = (k0 + 1 < Ksrc) ? W[(size_t)(k0 + 1) * Ncols + col] : 0.f;
    dst[tid] = pack2bf(w0, w1);
}

// ---------- pose features ----------
__global__ void pose_kernel(const float* __restrict__ lm,
                            float* __restrict__ poseOut,
                            unsigned short* __restrict__ comb) {
    int b = blockIdx.x * blockDim.x + threadIdx.x;
    if (b >= B_ROWS) return;
    const float* p = lm + (size_t)b * D_IN;
    float nx = p[1*3+0],  ny = p[1*3+1],  nz = p[1*3+2];
    float lex = p[33*3+0], ley = p[33*3+1], lez = p[33*3+2];
    float rex = p[263*3+0], rey = p[263*3+1], rez = p[263*3+2];
    float lmx = p[61*3+0],  rmx = p[291*3+0];
    float chy = p[18*3+1];
    float f[6];
    f[0] = nx - 0.5f * (lex + rex);
    f[1] = ny - 0.5f * (ley + rey);
    f[2] = nz - 0.5f * (lez + rez);
    f[3] = rex - lex;
    f[4] = rmx - lmx;
    f[5] = chy - ny;
    float* po = poseOut + (size_t)b * 6;
    unsigned short* cr = comb + (size_t)b * KCOMB;
    #pragma unroll
    for (int i = 0; i < 6; ++i) { po[i] = f[i]; cr[64 + i] = f2bf(f[i]); }
    #pragma unroll
    for (int i = 70; i < KCOMB; ++i) cr[i] = 0;
}

// ---------- generic WMMA bf16 GEMM: out = act(A @ W + bias) ----------
// block: 256 threads (8 wave32), 128 rows per block; NT n-tiles of 16, chunked by 8.
template<int NT, bool A_F32, bool RELU, bool OUT_BF, bool OUT_F>
__global__ __launch_bounds__(256)
void gemm_wmma_kernel(const void* __restrict__ Aptr, int lda, int Kreal, int KT,
                      const unsigned int* __restrict__ Wp,
                      const float* __restrict__ bias,
                      unsigned short* __restrict__ outB, int ldB,
                      float* __restrict__ outF, int ldF) {
    constexpr int CH     = (NT >= 8) ? 8 : NT;
    constexpr int CHUNKS = NT / CH;
    __shared__ __align__(16) unsigned short ldsA[128 * LDS_STRIDE];  // 128 rows x 32 bf16, padded

    const int rowBase = blockIdx.x * 128;
    const int lane = threadIdx.x & 31;
    const int wave = threadIdx.x >> 5;
    const int m    = lane & 15;
    const int half = lane >> 4;
    const float*          Af = (const float*)Aptr;
    const unsigned short* Ab = (const unsigned short*)Aptr;
    const v8f vzero = {0.f, 0.f, 0.f, 0.f, 0.f, 0.f, 0.f, 0.f};

    // staging coordinates (fixed per thread)
    const int sr  = threadIdx.x >> 1;          // row 0..127
    const int sc0 = (threadIdx.x & 1) << 4;    // col 0 or 16

    for (int chunk = 0; chunk < CHUNKS; ++chunk) {
        v8f acc[CH];
        #pragma unroll
        for (int c = 0; c < CH; ++c) acc[c] = vzero;

        for (int kt = 0; kt < KT; ++kt) {
            __syncthreads();
            // ---- stage 128x32 A panel (bf16) into LDS: wide loads + 2x ds_store_b128 ----
            {
                const int kbase = kt * 32 + sc0;
                unsigned int packed[8];
                if constexpr (A_F32) {
                    const float* srcRow = Af + (size_t)(rowBase + sr) * lda + kbase;
                    if (kbase + 16 <= Kreal) {
                        #pragma unroll
                        for (int j = 0; j < 4; ++j) {
                            float4 q = ((const float4*)srcRow)[j];
                            packed[2*j]   = pack2bf(q.x, q.y);
                            packed[2*j+1] = pack2bf(q.z, q.w);
                        }
                    } else {   // only the single partial tail tile takes this path
                        #pragma unroll
                        for (int j = 0; j < 8; ++j) {
                            int k0 = kbase + 2*j;
                            float f0 = (k0     < Kreal) ? srcRow[2*j]     : 0.f;
                            float f1 = (k0 + 1 < Kreal) ? srcRow[2*j + 1] : 0.f;
                            packed[j] = pack2bf(f0, f1);
                        }
                    }
                } else {       // bf16 activations are fully padded: unconditional copy
                    const uint4* srcRow = (const uint4*)(Ab + (size_t)(rowBase + sr) * lda + kbase);
                    *(uint4*)&packed[0] = srcRow[0];
                    *(uint4*)&packed[4] = srcRow[1];
                }
                uint4* dstRow = (uint4*)&ldsA[sr * LDS_STRIDE + sc0];
                dstRow[0] = *(uint4*)&packed[0];
                dstRow[1] = *(uint4*)&packed[4];
            }
            __syncthreads();
            // ---- gather A fragment (16-bit A 16x32 layout) ----
            union { v16bf v; unsigned int u[8]; } afrag;
            {
                int mrow = wave * 16 + m;
                #pragma unroll
                for (int v = 0; v < 8; ++v) {
                    int kb = 2 * v + 16 * half;
                    afrag.u[v] = *(const unsigned int*)&ldsA[mrow * LDS_STRIDE + kb];
                }
            }
            // ---- one WMMA per n-tile, B fragments streamed from L2 ----
            #pragma unroll
            for (int c = 0; c < CH; ++c) {
                int nt = chunk * CH + c;
                union { v16bf v; uint4 q[2]; } bfrag;
                const uint4* bp = (const uint4*)(Wp + (((size_t)kt * NT + nt) * 32 + lane) * 8);
                bfrag.q[0] = bp[0];
                bfrag.q[1] = bp[1];
                acc[c] = __builtin_amdgcn_wmma_f32_16x16x32_bf16(
                    false, afrag.v, false, bfrag.v, (short)0, acc[c], false, false);
            }
        }
        // ---- epilogue: C/D layout lane -> n = lane&15, rows = 8*half + r ----
        #pragma unroll
        for (int c = 0; c < CH; ++c) {
            int nt  = chunk * CH + c;
            int col = nt * 16 + m;
            float bv = bias[col];
            #pragma unroll
            for (int r = 0; r < 8; ++r) {
                int grow  = rowBase + wave * 16 + half * 8 + r;
                float val = acc[c][r] + bv;
                if constexpr (RELU) val = (val > 0.f) ? val : 0.f;
                if constexpr (OUT_F)  outF[(size_t)grow * ldF + col] = val;
                if constexpr (OUT_BF) outB[(size_t)grow * ldB + col] = f2bf(val);
            }
        }
    }
}

// ---------- classifier: out[b,c] = features[b,:] . Wc[:,c] + bc[c] ----------
__global__ void cls_kernel(const float* __restrict__ feat,
                           const float* __restrict__ Wc,
                           const float* __restrict__ bc,
                           float* __restrict__ out) {
    int gtid = blockIdx.x * blockDim.x + threadIdx.x;
    int row  = gtid >> 5;
    int lane = gtid & 31;
    if (row >= B_ROWS) return;
    const float* f = feat + (size_t)row * NF;
    float a0 = 0.f, a1 = 0.f;
    #pragma unroll
    for (int i = 0; i < NF / 32; ++i) {
        int k = lane + 32 * i;
        float v = f[k];
        a0 += v * Wc[2 * k + 0];
        a1 += v * Wc[2 * k + 1];
    }
    #pragma unroll
    for (int off = 16; off > 0; off >>= 1) {
        a0 += __shfl_down(a0, off);
        a1 += __shfl_down(a1, off);
    }
    if (lane == 0) {
        out[(size_t)row * 2 + 0] = a0 + bc[0];
        out[(size_t)row * 2 + 1] = a1 + bc[1];
    }
}

// ---------- host launch ----------
extern "C" void kernel_launch(void* const* d_in, const int* in_sizes, int n_in,
                              void* d_out, int out_size, void* d_ws, size_t ws_size,
                              hipStream_t stream) {
    const float* lm = (const float*)d_in[0];
    const float* W1 = (const float*)d_in[1];
    const float* b1 = (const float*)d_in[2];
    const float* W2 = (const float*)d_in[3];
    const float* b2 = (const float*)d_in[4];
    const float* W3 = (const float*)d_in[5];
    const float* b3 = (const float*)d_in[6];
    const float* Wf = (const float*)d_in[7];
    const float* bf = (const float*)d_in[8];
    const float* Wc = (const float*)d_in[9];
    const float* bc = (const float*)d_in[10];
    float* out = (float*)d_out;
    char*  ws  = (char*)d_ws;

    // d_out layout (floats): [output B*2][features B*512][pose B*6][lm_feat B*64]
    float* outO = out;
    float* outFeat = out + (size_t)B_ROWS * 2;
    float* outPose = out + (size_t)B_ROWS * (2 + NF);
    float* outLmf  = out + (size_t)B_ROWS * (2 + NF + 6);

    // workspace layout (bytes)
    const int KT1 = K1PAD / 32, NT1 = N1 / 16;       // 44, 16
    const int KT2 = 256 / 32,   NT2 = N2 / 16;       // 8, 8
    const int KT3 = 128 / 32,   NT3 = N3 / 16;       // 4, 4
    const int KT4 = KCOMB / 32, NT4 = NF / 16;       // 3, 32
    size_t off = 0;
    unsigned int* W1p = (unsigned int*)(ws + off); off += (size_t)KT1 * NT1 * 256 * 4;
    unsigned int* W2p = (unsigned int*)(ws + off); off += (size_t)KT2 * NT2 * 256 * 4;
    unsigned int* W3p = (unsigned int*)(ws + off); off += (size_t)KT3 * NT3 * 256 * 4;
    unsigned int* Wfp = (unsigned int*)(ws + off); off += (size_t)KT4 * NT4 * 256 * 4;
    unsigned short* h1   = (unsigned short*)(ws + off); off += (size_t)B_ROWS * N1 * 2;
    unsigned short* h2   = (unsigned short*)(ws + off); off += (size_t)B_ROWS * N2 * 2;
    unsigned short* comb = (unsigned short*)(ws + off); off += (size_t)B_ROWS * KCOMB * 2;
    (void)ws_size; (void)in_sizes; (void)n_in; (void)out_size;

    // 1) pack weights into WMMA B-fragment layout (bf16)
    pack_w_kernel<<<dim3((KT1 * NT1 * 256 + 255) / 256), dim3(256), 0, stream>>>(W1, D_IN, N1, KT1, NT1, W1p);
    pack_w_kernel<<<dim3((KT2 * NT2 * 256 + 255) / 256), dim3(256), 0, stream>>>(W2, 256,  N2, KT2, NT2, W2p);
    pack_w_kernel<<<dim3((KT3 * NT3 * 256 + 255) / 256), dim3(256), 0, stream>>>(W3, 128,  N3, KT3, NT3, W3p);
    pack_w_kernel<<<dim3((KT4 * NT4 * 256 + 255) / 256), dim3(256), 0, stream>>>(Wf, 70,   NF, KT4, NT4, Wfp);

    // 2) pose features (also fills cols 64..95 of the concat buffer)
    pose_kernel<<<dim3(B_ROWS / 256), dim3(256), 0, stream>>>(lm, outPose, comb);

    const int gemmBlocks = B_ROWS / 128;
    // 3) L1: x(f32,1404) @ W1 -> relu -> h1 bf16
    gemm_wmma_kernel<NT1, true,  true,  true,  false><<<dim3(gemmBlocks), dim3(256), 0, stream>>>(
        lm, D_IN, D_IN, KT1, W1p, b1, h1, N1, nullptr, 0);
    // 4) L2: h1 @ W2 -> relu -> h2 bf16
    gemm_wmma_kernel<NT2, false, true,  true,  false><<<dim3(gemmBlocks), dim3(256), 0, stream>>>(
        h1, N1, 256, KT2, W2p, b2, h2, N2, nullptr, 0);
    // 5) L3: h2 @ W3 -> lm_feat (f32 out + bf16 into comb cols 0..63)
    gemm_wmma_kernel<NT3, false, false, true,  true ><<<dim3(gemmBlocks), dim3(256), 0, stream>>>(
        h2, N2, 128, KT3, W3p, b3, comb, KCOMB, outLmf, N3);
    // 6) L4: comb @ Wf -> relu -> features f32
    gemm_wmma_kernel<NT4, false, true,  false, true ><<<dim3(gemmBlocks), dim3(256), 0, stream>>>(
        comb, KCOMB, KCOMB, KT4, Wfp, bf, nullptr, 0, outFeat, NF);
    // 7) classifier: features @ Wc + bc
    cls_kernel<<<dim3(B_ROWS / 8), dim3(256), 0, stream>>>(outFeat, Wc, bc, outO);
}
